// ColorEncoder_31894427140212
// MI455X (gfx1250) — matrix operand order repaired and checked by reading, not compiled, not run on hardware
//
#include <hip/hip_runtime.h>
#include <hip/hip_bf16.h>
#include <math.h>

typedef __attribute__((ext_vector_type(16))) _Float16 v16h;
typedef __attribute__((ext_vector_type(8)))  float    v8f;
typedef int i32x4 __attribute__((vector_size(16)));

#define EPSF 1e-5f

// ---------------- CDNA5 async global->LDS helpers --------------------------
#if defined(__has_builtin)
#if __has_builtin(__builtin_amdgcn_global_load_async_to_lds_b128)
#define HAVE_ASYNC_LDS_BUILTIN 1
#endif
#if __has_builtin(__builtin_amdgcn_s_wait_asynccnt)
#define HAVE_ASYNC_WAIT_BUILTIN 1
#endif
#endif

__device__ __forceinline__ void async_copy_b128(_Float16* lds_dst,
                                                const _Float16* gsrc) {
#ifdef HAVE_ASYNC_LDS_BUILTIN
  __builtin_amdgcn_global_load_async_to_lds_b128(
      (__attribute__((address_space(1))) i32x4*)gsrc,
      (__attribute__((address_space(3))) i32x4*)lds_dst, 0, 0);
#else
  unsigned lo =
      (unsigned)(size_t)((__attribute__((address_space(3))) void*)lds_dst);
  asm volatile("global_load_async_to_lds_b128 %0, %1, off"
               :
               : "v"(lo), "v"(gsrc)
               : "memory");
#endif
}

__device__ __forceinline__ void wait_async_le2() {
#ifdef HAVE_ASYNC_WAIT_BUILTIN
  __builtin_amdgcn_s_wait_asynccnt(2);
#else
  asm volatile("s_wait_asynccnt 0x2" ::: "memory");
#endif
}

__device__ __forceinline__ void wait_async_0() {
#ifdef HAVE_ASYNC_WAIT_BUILTIN
  __builtin_amdgcn_s_wait_asynccnt(0);
#else
  asm volatile("s_wait_asynccnt 0x0" ::: "memory");
#endif
}

// ---------------- WMMA fragment helpers (CDNA5 16x16x32 f16, wave32) -------
// A matrix 16x32 (MxK) f16: lanes 0-15 row m=lane hold K=0..15,
// lanes 16-31 row m=lane-16 hold K=16..31; 2 f16 per VGPR.
__device__ __forceinline__ v16h load_a_frag(const _Float16* A, int lda) {
  const int lane = threadIdx.x & 31;
  const int m = lane & 15;
  const int kh = (lane >> 4) << 4;
  v16h a;
#pragma unroll
  for (int v = 0; v < 8; ++v) {
    int k = 2 * v + kh;
    a[2 * v]     = A[m * lda + k];
    a[2 * v + 1] = A[m * lda + k + 1];
  }
  return a;
}

// B matrix 32x16 (KxN); lane holds column n=lane&15, K range 16*(lane>>4)..+15.
// Source stored [N][K] row-major (contiguous K per column) -> 32B/lane load.
__device__ __forceinline__ v16h load_b_frag_nk(const _Float16* B, int ldb) {
  const int lane = threadIdx.x & 31;
  const int n = lane & 15;
  const int k0 = (lane >> 4) << 4;
  v16h b;
#pragma unroll
  for (int v = 0; v < 16; ++v) b[v] = B[n * ldb + k0 + v];
  return b;
}

// Same fragment, source stored [K][N] row-major.
__device__ __forceinline__ v16h load_b_frag_kn(const _Float16* B, int ldb) {
  const int lane = threadIdx.x & 31;
  const int n = lane & 15;
  const int k0 = (lane >> 4) << 4;
  v16h b;
#pragma unroll
  for (int v = 0; v < 16; ++v) b[v] = B[(k0 + v) * ldb + n];
  return b;
}

__device__ __forceinline__ v8f wmma_f16(v16h a, v16h b, v8f c) {
  return __builtin_amdgcn_wmma_f32_16x16x32_f16(false, a, false, b, (short)0, c,
                                                false, false);
}

__device__ __forceinline__ float gelu_exact(float x) {
  return 0.5f * x * (1.f + erff(x * 0.70710678118654752f));
}

// ---------------- conv1 (3->128, 3x3 s2 p1) + BN + ReLU -> f16 NHWC --------
__global__ __launch_bounds__(256) void conv1_kernel(
    const float* __restrict__ img, const float* __restrict__ w1,
    const float* __restrict__ cb, const float* __restrict__ g,
    const float* __restrict__ be, const float* __restrict__ mm,
    const float* __restrict__ vv, _Float16* __restrict__ feat1) {
  int gid = blockIdx.x * 256 + threadIdx.x;
  int co = gid & 127;
  int p = gid >> 7;
  int x = p & 127; p >>= 7;
  int y = p & 127; p >>= 7;
  int b = p;
  float scale = g[co] * rsqrtf(vv[co] + EPSF);
  float bias = (cb[co] - mm[co]) * scale + be[co];
  float acc = 0.f;
#pragma unroll
  for (int ci = 0; ci < 3; ++ci)
#pragma unroll
    for (int ky = 0; ky < 3; ++ky) {
      int iy = 2 * y - 1 + ky;
      if (iy < 0 || iy > 255) continue;
#pragma unroll
      for (int kx = 0; kx < 3; ++kx) {
        int ix = 2 * x - 1 + kx;
        if (ix < 0 || ix > 255) continue;
        acc += img[((b * 3 + ci) * 256 + iy) * 256 + ix] *
               w1[((co * 3 + ci) * 3 + ky) * 3 + kx];
      }
    }
  float o = fmaxf(acc * scale + bias, 0.f);
  feat1[((b * 128 + y) * 128 + x) * 128 + co] = (_Float16)o;
}

// ---------------- weight packing ------------------------------------------
// w2h[co][pos][c] = conv2_w[co][c][pos] (pos = ky*3+kx), f16
__global__ void pack_w2_kernel(const float* __restrict__ w2,
                               _Float16* __restrict__ w2h) {
  int gidx = blockIdx.x * 256 + threadIdx.x;
  if (gidx >= 128 * 1152) return;
  int co = gidx / 1152;
  int r = gidx - co * 1152;
  int pos = r >> 7;
  int c = r & 127;
  w2h[gidx] = (_Float16)w2[(co * 128 + c) * 9 + pos];
}

__global__ void pack_valw_kernel(const float* __restrict__ w,
                                 _Float16* __restrict__ wh) {
  int gidx = blockIdx.x * 256 + threadIdx.x;
  if (gidx < 128 * 128) wh[gidx] = (_Float16)w[gidx];
}

// Wq[q][c'] = sum_c cq[q][c]*key_w[c][c'];  bq[q] = cq[q]·key_b
__global__ __launch_bounds__(128) void prep_q_kernel(
    const float* __restrict__ cq, const float* __restrict__ key_w,
    const float* __restrict__ key_b, _Float16* __restrict__ Wq,
    float* __restrict__ bq) {
  int q = blockIdx.x;
  int cp = threadIdx.x;
  float s = 0.f;
  for (int c = 0; c < 128; ++c) s += cq[q * 128 + c] * key_w[c * 128 + cp];
  Wq[q * 128 + cp] = (_Float16)s;
  if (cp == 0) {
    float t = 0.f;
    for (int c = 0; c < 128; ++c) t += cq[q * 128 + c] * key_b[c];
    bq[q] = t;
  }
}

// ---------------- conv2 (128->128, 3x3 s2 p1) implicit GEMM via WMMA -------
// M=128 (co), N=32 pixels per block, K=1152 in 36 chunks of 32 (one (ky,kx),
// 32-channel slice per chunk). Weight tile staged via async-to-LDS DMA.
// Fused BN+ReLU. Output fp32 NHWC [B][64][64][128].
__global__ __launch_bounds__(256) void conv2_kernel(
    const _Float16* __restrict__ feat1, const _Float16* __restrict__ w2h,
    const float* __restrict__ cb, const float* __restrict__ g2,
    const float* __restrict__ b2, const float* __restrict__ m2,
    const float* __restrict__ v2, float* __restrict__ feat2) {
  __shared__ __align__(16) _Float16 As[128 * 32];  // [co][k]
  __shared__ __align__(16) _Float16 Bs[32 * 32];   // [pixel][k]
  __shared__ float s_scale[128];
  __shared__ float s_bias[128];
  const int tid = threadIdx.x;
  int blk = blockIdx.x;
  int b = blk >> 7;
  int rem = blk & 127;
  int y = rem >> 1;
  int x0 = (rem & 1) * 32;

  if (tid < 128) {
    float sc = g2[tid] * rsqrtf(v2[tid] + EPSF);
    s_scale[tid] = sc;
    s_bias[tid] = (cb[tid] - m2[tid]) * sc + b2[tid];
  }

  const int wid = tid >> 5;
  v8f acc0 = {}, acc1 = {};

  for (int kc = 0; kc < 36; ++kc) {
    int pos = kc >> 2;
    int cb0 = (kc & 3) << 5;
    int ky = pos / 3, kx = pos - ky * 3;
    {  // stage A via async DMA: 128 rows x 32 k, 2x b128 per thread
      int co = tid >> 1;
      int j0 = (tid & 1) << 4;
      const _Float16* src = w2h + co * 1152 + pos * 128 + cb0 + j0;
      _Float16* dst = As + co * 32 + j0;
      async_copy_b128(dst, src);
      async_copy_b128(dst + 8, src + 8);
    }
    {  // stage B: 32 pixels x 32 channels (im2col gather, zero-padded)
      int n = tid >> 3;
      int c4 = (tid & 7) << 2;
      int iy = 2 * y - 1 + ky;
      int ix = 2 * (x0 + n) - 1 + kx;
      _Float16* dst = Bs + n * 32 + c4;
      if (iy >= 0 && iy < 128 && ix >= 0 && ix < 128) {
        const _Float16* src =
            feat1 + (((b * 128 + iy) * 128 + ix) * 128 + cb0 + c4);
#pragma unroll
        for (int j = 0; j < 4; ++j) dst[j] = src[j];
      } else {
#pragma unroll
        for (int j = 0; j < 4; ++j) dst[j] = (_Float16)0.f;
      }
      int kcn = kc + 1;
      if (kcn < 36) {  // prefetch next chunk's im2col source row
        int posn = kcn >> 2, cbn = (kcn & 3) << 5;
        int kyn = posn / 3, kxn = posn - kyn * 3;
        int iyn = 2 * y - 1 + kyn;
        int ixn = 2 * (x0 + n) - 1 + kxn;
        if (iyn >= 0 && iyn < 128 && ixn >= 0 && ixn < 128)
          __builtin_prefetch(
              feat1 + (((b * 128 + iyn) * 128 + ixn) * 128 + cbn + c4), 0, 1);
      }
    }
    wait_async_0();
    __syncthreads();
    v16h a = load_a_frag(As + wid * 16 * 32, 32);
    v16h bb0 = load_b_frag_nk(Bs, 32);
    v16h bb1 = load_b_frag_nk(Bs + 16 * 32, 32);
    acc0 = wmma_f16(a, bb0, acc0);
    acc1 = wmma_f16(a, bb1, acc1);
    __syncthreads();
  }
  const int lane = tid & 31;
  const int nlo = lane & 15;
  const int mb = wid * 16 + ((lane >> 4) << 3);
#pragma unroll
  for (int v = 0; v < 8; ++v) {
    int m = mb + v;
    float sc = s_scale[m], bi = s_bias[m];
    float o0 = fmaxf(acc0[v] * sc + bi, 0.f);
    float o1 = fmaxf(acc1[v] * sc + bi, 0.f);
    feat2[((b * 64 + y) * 64 + x0 + nlo) * 128 + m] = o0;
    feat2[((b * 64 + y) * 64 + x0 + 16 + nlo) * 128 + m] = o1;
  }
}

// ---------------- depthwise 3x3 + residual + LayerNorm -> tln f16 ---------
__global__ __launch_bounds__(128) void dw_ln_kernel(
    const float* __restrict__ feat2, const float* __restrict__ dw_w,
    const float* __restrict__ dw_b, const float* __restrict__ g1,
    const float* __restrict__ b1, _Float16* __restrict__ tln) {
  __shared__ float red[128];
  int blk = blockIdx.x;
  int b = blk >> 12;
  int p = blk & 4095;
  int y = p >> 6, x = p & 63;
  int c = threadIdx.x;
  float acc = dw_b[c] + feat2[((b * 64 + y) * 64 + x) * 128 + c];
#pragma unroll
  for (int ky = 0; ky < 3; ++ky) {
    int yy = y + ky - 1;
    if (yy < 0 || yy > 63) continue;
#pragma unroll
    for (int kx = 0; kx < 3; ++kx) {
      int xx = x + kx - 1;
      if (xx < 0 || xx > 63) continue;
      acc += feat2[((b * 64 + yy) * 64 + xx) * 128 + c] *
             dw_w[c * 9 + ky * 3 + kx];
    }
  }
  red[c] = acc; __syncthreads();
  for (int s = 64; s > 0; s >>= 1) { if (c < s) red[c] += red[c + s]; __syncthreads(); }
  float mu = red[0] * (1.f / 128.f); __syncthreads();
  float d = acc - mu;
  red[c] = d * d; __syncthreads();
  for (int s = 64; s > 0; s >>= 1) { if (c < s) red[c] += red[c + s]; __syncthreads(); }
  float var = red[0] * (1.f / 128.f);
  tln[(b * 4096 + p) * 128 + c] = (_Float16)(d * rsqrtf(var + EPSF) * g1[c] + b1[c]);
}

// ---------------- m_all[b][q][t] = tln·Wq^T + bq  (WMMA, K=128) ------------
__global__ __launch_bounds__(256) void mall_kernel(
    const _Float16* __restrict__ tln, const _Float16* __restrict__ Wq,
    const float* __restrict__ bq, float* __restrict__ m_all) {
  int blk = blockIdx.x;            // b*512 + mb*256 + tb
  int b = blk >> 9;
  int rem = blk & 511;
  int m0 = (rem >> 8) << 7;
  int t0 = (rem & 255) << 4;
  int wid = threadIdx.x >> 5;
  int j0 = m0 + wid * 16;
  v8f acc = {};
#pragma unroll
  for (int kc = 0; kc < 4; ++kc) {
    v16h a = load_a_frag(Wq + j0 * 128 + kc * 32, 128);
    v16h bb = load_b_frag_nk(tln + (b * 4096 + t0) * 128 + kc * 32, 128);
    acc = wmma_f16(a, bb, acc);
  }
  int lane = threadIdx.x & 31;
  int t = t0 + (lane & 15);
  int jb = j0 + ((lane >> 4) << 3);
#pragma unroll
  for (int v = 0; v < 8; ++v)
    m_all[(jb + v + b * 256) * 4096 + t] = acc[v] + bq[jb + v];
}

// ---------------- v[b][t][c] = tln·val_w^T + val_b  (WMMA, K=128) ----------
__global__ __launch_bounds__(256) void vproj_kernel(
    const _Float16* __restrict__ tln, const _Float16* __restrict__ Wv,
    const float* __restrict__ vb, _Float16* __restrict__ vbuf) {
  int blk = blockIdx.x;            // b*256 + tb
  int b = blk >> 8;
  int t0 = (blk & 255) << 4;
  int wid = threadIdx.x >> 5;
  int c0 = wid * 16;
  v8f acc = {};
#pragma unroll
  for (int kc = 0; kc < 4; ++kc) {
    v16h a = load_a_frag(Wv + c0 * 128 + kc * 32, 128);
    v16h bb = load_b_frag_nk(tln + (b * 4096 + t0) * 128 + kc * 32, 128);
    acc = wmma_f16(a, bb, acc);
  }
  int lane = threadIdx.x & 31;
  int t = t0 + (lane & 15);
  int cbs = c0 + ((lane >> 4) << 3);
#pragma unroll
  for (int v = 0; v < 8; ++v)
    vbuf[(b * 4096 + t) * 128 + cbs + v] = (_Float16)(acc[v] + vb[cbs + v]);
}

// ---------------- squared norms of m_all rows ------------------------------
__global__ __launch_bounds__(256) void norms_kernel(
    const float* __restrict__ m_all, float* __restrict__ normsq) {
  __shared__ float red[256];
  const float* src = m_all + (size_t)blockIdx.x * 4096;
  float s = 0.f;
  for (int j = threadIdx.x; j < 4096; j += 256) { float v = src[j]; s += v * v; }
  red[threadIdx.x] = s; __syncthreads();
  for (int st = 128; st > 0; st >>= 1) {
    if (threadIdx.x < st) red[threadIdx.x] += red[threadIdx.x + st];
    __syncthreads();
  }
  if (threadIdx.x == 0) normsq[blockIdx.x] = red[0];
}

// ---------------- top-64 of 256 per batch (rank counting, stable desc) -----
__global__ __launch_bounds__(256) void topk_kernel(
    const float* __restrict__ normsq, int* __restrict__ idx) {
  __shared__ float sn[256];
  int b = blockIdx.x;
  int q = threadIdx.x;
  sn[q] = normsq[b * 256 + q];
  __syncthreads();
  float mv = sn[q];
  int rank = 0;
  for (int j = 0; j < 256; ++j) {
    float o = sn[j];
    rank += (o > mv) || (o == mv && j < q);
  }
  if (rank < 64) idx[b * 64 + rank] = q;
}

// ---------------- softmax over HW + write m_log + attn(f16) ----------------
__global__ __launch_bounds__(256) void softmax_kernel(
    const float* __restrict__ m_all, const int* __restrict__ idx,
    _Float16* __restrict__ attn, float* __restrict__ mlog_out) {
  __shared__ float red[256];
  int blk = blockIdx.x;            // b*64 + i
  int b = blk >> 6;
  int q = idx[blk];
  const float* row = m_all + (size_t)(b * 256 + q) * 4096;
  int tid = threadIdx.x;
  float loc[16];
  float lmax = -1e30f;
#pragma unroll
  for (int j = 0; j < 16; ++j) {
    loc[j] = row[tid + j * 256];
    lmax = fmaxf(lmax, loc[j]);
  }
  red[tid] = lmax; __syncthreads();
  for (int st = 128; st > 0; st >>= 1) {
    if (tid < st) red[tid] = fmaxf(red[tid], red[tid + st]);
    __syncthreads();
  }
  float mx = red[0]; __syncthreads();
  const float s = 0.08838834764831845f;  // 1/sqrt(128)
  float lsum = 0.f;
#pragma unroll
  for (int j = 0; j < 16; ++j) {
    mlog_out[(size_t)blk * 4096 + tid + j * 256] = logf(loc[j]);
    loc[j] = expf((loc[j] - mx) * s);
    lsum += loc[j];
  }
  red[tid] = lsum; __syncthreads();
  for (int st = 128; st > 0; st >>= 1) {
    if (tid < st) red[tid] += red[tid + st];
    __syncthreads();
  }
  float inv = 1.f / red[0];
#pragma unroll
  for (int j = 0; j < 16; ++j)
    attn[(size_t)blk * 4096 + tid + j * 256] = (_Float16)(loc[j] * inv);
}

// ---------------- x = attn @ v  (per batch: 64x4096 @ 4096x128, WMMA) ------
// Double-buffered LDS with async global->LDS DMA staging of v tiles.
__global__ __launch_bounds__(256) void attnv_kernel(
    const _Float16* __restrict__ attn, const _Float16* __restrict__ vbuf,
    float* __restrict__ xbuf) {
  __shared__ __align__(16) _Float16 Vs[2][32 * 128];  // [k][c]
  int b = blockIdx.x;
  int tid = threadIdx.x;
  int wid = tid >> 5;
  int mtile = wid >> 1;
  int ntb = (wid & 1) * 4;
  int k = tid >> 3;
  int c0 = (tid & 7) << 4;
  const _Float16* srcbase = vbuf + (size_t)(b * 4096 + k) * 128 + c0;
  const size_t chunk = (size_t)32 * 128;

  // prologue: stage chunk 0 into buffer 0
  async_copy_b128(&Vs[0][k * 128 + c0], srcbase);
  async_copy_b128(&Vs[0][k * 128 + c0 + 8], srcbase + 8);

  v8f acc[4] = {};
  for (int kc = 0; kc < 128; ++kc) {
    int buf = kc & 1;
    if (kc + 1 < 128) {  // stage next chunk into other buffer
      const _Float16* s2 = srcbase + (size_t)(kc + 1) * chunk;
      async_copy_b128(&Vs[buf ^ 1][k * 128 + c0], s2);
      async_copy_b128(&Vs[buf ^ 1][k * 128 + c0 + 8], s2 + 8);
      wait_async_le2();  // chunk kc's 2 ops complete (in-order)
    } else {
      wait_async_0();
    }
    __syncthreads();
    v16h a =
        load_a_frag(attn + (size_t)(b * 64 + mtile * 16) * 4096 + kc * 32, 4096);
#pragma unroll
    for (int u = 0; u < 4; ++u) {
      v16h bb = load_b_frag_kn(&Vs[buf][(ntb + u) * 16], 128);
      acc[u] = wmma_f16(a, bb, acc[u]);
    }
    __syncthreads();
  }
  int lane = tid & 31;
  int i = mtile * 16 + ((lane >> 4) << 3);
  int n = lane & 15;
#pragma unroll
  for (int u = 0; u < 4; ++u) {
    int c = (ntb + u) * 16 + n;
#pragma unroll
    for (int v = 0; v < 8; ++v)
      xbuf[(size_t)(b * 64 + i + v) * 128 + c] = acc[u][v];
  }
}

// ---------------- out-proj, LN2, MLP, residual, color head -----------------
__global__ __launch_bounds__(128) void tail_kernel(
    const float* __restrict__ xbuf,
    const float* __restrict__ out_w, const float* __restrict__ out_b,
    const float* __restrict__ ln2_g, const float* __restrict__ ln2_b,
    const float* __restrict__ mlp_w1, const float* __restrict__ mlp_b1,
    const float* __restrict__ mlp_w2, const float* __restrict__ mlp_b2,
    const float* __restrict__ ch_w1, const float* __restrict__ ch_b1,
    const float* __restrict__ ch_w2, const float* __restrict__ ch_b2,
    float* __restrict__ palette) {
  __shared__ float sA[128], sB[128], sC[128], red[128];
  int row = blockIdx.x;   // b*64 + i
  int c = threadIdx.x;
  sA[c] = xbuf[row * 128 + c];
  __syncthreads();
  float xo = out_b[c];
  for (int k = 0; k < 128; ++k) xo += out_w[c * 128 + k] * sA[k];
  sB[c] = xo;
  __syncthreads();
  // LN2
  red[c] = xo; __syncthreads();
  for (int s = 64; s > 0; s >>= 1) { if (c < s) red[c] += red[c + s]; __syncthreads(); }
  float mu = red[0] * (1.f / 128.f); __syncthreads();
  float d = xo - mu;
  red[c] = d * d; __syncthreads();
  for (int s = 64; s > 0; s >>= 1) { if (c < s) red[c] += red[c + s]; __syncthreads(); }
  float var = red[0] * (1.f / 128.f); __syncthreads();
  sA[c] = d * rsqrtf(var + EPSF) * ln2_g[c] + ln2_b[c];
  __syncthreads();
  // MLP
  float m1 = mlp_b1[c];
  for (int k = 0; k < 128; ++k) m1 += mlp_w1[c * 128 + k] * sA[k];
  sC[c] = gelu_exact(m1);
  __syncthreads();
  float y = mlp_b2[c];
  for (int k = 0; k < 128; ++k) y += mlp_w2[c * 128 + k] * sC[k];
  float h = sB[c] + y;
  __syncthreads();
  sA[c] = h;
  __syncthreads();
  // color head
  float c1 = ch_b1[c];
  for (int k = 0; k < 128; ++k) c1 += ch_w1[c * 128 + k] * sA[k];
  sB[c] = gelu_exact(c1);
  __syncthreads();
  if (c < 3) {
    float p = ch_b2[c];
    for (int k = 0; k < 128; ++k) p += ch_w2[c * 128 + k] * sB[k];
    p = 1.f / (1.f + expf(-p));
    int b = row >> 6, i = row & 63;
    palette[(b * 3 + c) * 64 + i] = p;   // [B,3,nc] transposed output
  }
}

// ---------------- host launch ---------------------------------------------
extern "C" void kernel_launch(void* const* d_in, const int* in_sizes, int n_in,
                              void* d_out, int out_size, void* d_ws,
                              size_t ws_size, hipStream_t stream) {
  (void)in_sizes; (void)n_in; (void)out_size; (void)ws_size;
  const float* img     = (const float*)d_in[0];
  const float* cq      = (const float*)d_in[1];
  // d_in[2] = num_colors (compile-time 64)
  const float* conv1_w = (const float*)d_in[3];
  const float* conv1_b = (const float*)d_in[4];
  const float* bn1_g   = (const float*)d_in[5];
  const float* bn1_b   = (const float*)d_in[6];
  const float* bn1_m   = (const float*)d_in[7];
  const float* bn1_v   = (const float*)d_in[8];
  const float* conv2_w = (const float*)d_in[9];
  const float* conv2_b = (const float*)d_in[10];
  const float* bn2_g   = (const float*)d_in[11];
  const float* bn2_b   = (const float*)d_in[12];
  const float* bn2_m   = (const float*)d_in[13];
  const float* bn2_v   = (const float*)d_in[14];
  const float* dw_w    = (const float*)d_in[15];
  const float* dw_b    = (const float*)d_in[16];
  const float* ln1_g   = (const float*)d_in[17];
  const float* ln1_b   = (const float*)d_in[18];
  const float* key_w   = (const float*)d_in[19];
  const float* key_b   = (const float*)d_in[20];
  const float* val_w   = (const float*)d_in[21];
  const float* val_b   = (const float*)d_in[22];
  const float* out_w   = (const float*)d_in[23];
  const float* out_b   = (const float*)d_in[24];
  const float* ln2_g   = (const float*)d_in[25];
  const float* ln2_b   = (const float*)d_in[26];
  const float* mlp_w1  = (const float*)d_in[27];
  const float* mlp_b1  = (const float*)d_in[28];
  const float* mlp_w2  = (const float*)d_in[29];
  const float* mlp_b2  = (const float*)d_in[30];
  const float* ch_w1   = (const float*)d_in[31];
  const float* ch_b1   = (const float*)d_in[32];
  const float* ch_w2   = (const float*)d_in[33];
  const float* ch_b2   = (const float*)d_in[34];

  char* ws = (char*)d_ws;
  // feat1 (f16, 67.1MB) is dead after conv2 -> reused as m_all (f32, 67.1MB)
  _Float16* feat1  = (_Float16*)(ws);
  float*    m_all  = (float*)(ws);
  // feat2 (f32, 33.5MB) is dead after dw_ln -> reused for v/attn/x/norm/idx
  float*    feat2  = (float*)(ws + (size_t)67108864);
  _Float16* vbuf   = (_Float16*)(ws + (size_t)67108864);   // 16.8MB
  _Float16* attn   = (_Float16*)(ws + (size_t)83886080);   // 8.4MB
  float*    xbuf   = (float*)(ws + (size_t)92274688);      // 0.5MB
  float*    normsq = (float*)(ws + (size_t)92798976);      // 16KB
  int*      idx    = (int*)(ws + (size_t)92815360);        // 4KB
  _Float16* tln    = (_Float16*)(ws + (size_t)100663296);  // 16.8MB
  _Float16* w2h    = (_Float16*)(ws + (size_t)117440512);  // 288KB
  _Float16* Wq     = (_Float16*)(ws + (size_t)117735424);  // 64KB
  float*    bq     = (float*)(ws + (size_t)117800960);     // 1KB
  _Float16* Wv     = (_Float16*)(ws + (size_t)117805056);  // 32KB

  float* out_palette = (float*)d_out;          // 16*3*64
  float* out_mlog    = (float*)d_out + 3072;   // 16*64*64*64

  conv1_kernel<<<131072, 256, 0, stream>>>(img, conv1_w, conv1_b, bn1_g, bn1_b,
                                           bn1_m, bn1_v, feat1);
  pack_w2_kernel<<<576, 256, 0, stream>>>(conv2_w, w2h);
  pack_valw_kernel<<<64, 256, 0, stream>>>(val_w, Wv);
  prep_q_kernel<<<256, 128, 0, stream>>>(cq, key_w, key_b, Wq, bq);
  conv2_kernel<<<2048, 256, 0, stream>>>(feat1, w2h, conv2_b, bn2_g, bn2_b,
                                         bn2_m, bn2_v, feat2);
  dw_ln_kernel<<<65536, 128, 0, stream>>>(feat2, dw_w, dw_b, ln1_g, ln1_b, tln);
  vproj_kernel<<<4096, 256, 0, stream>>>(tln, Wv, val_b, vbuf);  // feat2 dead
  mall_kernel<<<8192, 256, 0, stream>>>(tln, Wq, bq, m_all);     // feat1 dead
  norms_kernel<<<4096, 256, 0, stream>>>(m_all, normsq);
  topk_kernel<<<16, 256, 0, stream>>>(normsq, idx);
  softmax_kernel<<<1024, 256, 0, stream>>>(m_all, idx, attn, out_mlog);
  attnv_kernel<<<16, 256, 0, stream>>>(attn, vbuf, xbuf);
  tail_kernel<<<1024, 128, 0, stream>>>(xbuf, out_w, out_b, ln2_g, ln2_b,
                                        mlp_w1, mlp_b1, mlp_w2, mlp_b2, ch_w1,
                                        ch_b1, ch_w2, ch_b2, out_palette);
}
